// GNN_79937931313413
// MI455X (gfx1250) — matrix operand order repaired and checked by reading, not compiled, not run on hardware
//
#include <hip/hip_runtime.h>

// ---------- types ----------
typedef __attribute__((ext_vector_type(16))) __bf16        v16bf;
typedef __attribute__((ext_vector_type(8)))  float         v8f;
typedef __attribute__((ext_vector_type(4)))  unsigned int  u32x4;

union Frag16 { v16bf v; u32x4 q[2]; };

// ---------- helpers ----------
__device__ __forceinline__ unsigned short f2bf(float f) {
    unsigned u = __float_as_uint(f);
    unsigned r = u + 0x7fffu + ((u >> 16) & 1u);   // round-to-nearest-even
    return (unsigned short)(r >> 16);
}
// monotone float<->uint encoding for atomicMax over signed floats
__device__ __forceinline__ unsigned encf(float f) {
    unsigned u = __float_as_uint(f);
    return (u & 0x80000000u) ? ~u : (u | 0x80000000u);
}
__device__ __forceinline__ float decf(unsigned u) {
    u = (u & 0x80000000u) ? (u & 0x7fffffffu) : ~u;
    return __uint_as_float(u);
}

// ---------- weight pre-pack into WMMA B-fragment layout ----------
// WP index = ((mat*8 + ntile)*4 + kc)*32*16 + lane*16 + j
// value    = W[kc*32 + (lane>>4)*16 + j][ntile*16 + (lane&15)]
__global__ void pack_w_k(const float* __restrict__ W0, const float* __restrict__ W1,
                         const float* __restrict__ W2, const float* __restrict__ W3,
                         unsigned short* __restrict__ wp) {
    int idx  = blockIdx.x * 256 + threadIdx.x;      // 4*128*128 = 65536 total
    int mat  = idx >> 14;
    int rem  = idx & 16383;
    int nt   = rem >> 11;
    int kc   = (rem >> 9) & 3;
    int lane = (rem >> 4) & 31;
    int j    = rem & 15;
    int k    = kc * 32 + (lane >> 4) * 16 + j;
    int col  = nt * 16 + (lane & 15);
    const float* W = (mat == 0) ? W0 : (mat == 1) ? W1 : (mat == 2) ? W2 : W3;
    wp[idx] = f2bf(W[k * 128 + col]);
}

// ---------- f32 -> bf16 conversion ----------
__global__ void cvt_bf16_k(const float* __restrict__ in, unsigned short* __restrict__ out, int n) {
    int i = blockIdx.x * blockDim.x + threadIdx.x;
    if (i < n) out[i] = f2bf(in[i]);
}

// ---------- WMMA GEMM: OUT[matsel] = XB @ W[matBase+matsel]  (N x 128) ----------
__global__ void __launch_bounds__(256)
gemm_bf16_k(const unsigned short* __restrict__ XB, const unsigned short* __restrict__ WP,
            float* __restrict__ out0, float* __restrict__ out1, int matBase, int nrows) {
    const int wave = threadIdx.x >> 5;            // 0..7 -> N-tile
    const int lane = threadIdx.x & 31;
    const int half = lane >> 4;
    const int n16  = lane & 15;
    const int tile = blockIdx.x;                  // M-tile (16 rows)
    const int mat  = matBase + blockIdx.y;
    float* __restrict__ out = blockIdx.y ? out1 : out0;

    int arow = tile * 16 + n16;                   // A row this lane holds
    if (arow >= nrows) arow = nrows - 1;
    const unsigned short* __restrict__ ap = XB + (size_t)arow * 128;

    v8f c = {0.f, 0.f, 0.f, 0.f, 0.f, 0.f, 0.f, 0.f};
#pragma unroll
    for (int kc = 0; kc < 4; ++kc) {
        Frag16 a, b;
        // A fragment in this code's own k order: per lane one contiguous 16-half run (k = 16*half + s), matching packed B
        a.q[0] = *(const u32x4*)(ap + kc * 32 + half * 16);
        a.q[1] = *(const u32x4*)(ap + kc * 32 + half * 16 + 8);
        // pre-packed B: 32 contiguous bytes per lane
        const unsigned short* __restrict__ bp =
            WP + ((((size_t)mat * 8 + wave) * 4 + kc) * 32 + lane) * 16;
        b.q[0] = *(const u32x4*)(bp);
        b.q[1] = *(const u32x4*)(bp + 8);
        c = __builtin_amdgcn_wmma_f32_16x16x32_bf16(false, a.v, false, b.v,
                                                    (short)0, c, false, false);
    }
    // C/D layout: VGPR r -> row (rbase + r), col = n16 within the wave's N-tile
    const int col   = wave * 16 + n16;
    const int rbase = tile * 16 + half * 8;
    float* __restrict__ op = out + (size_t)rbase * 128 + col;
    if (tile * 16 + 16 <= nrows) {
        // uniform fast path: whole tile in range, no per-lane guards
#pragma unroll
        for (int r = 0; r < 8; ++r) op[(size_t)r * 128] = c[r];
    } else {
#pragma unroll
        for (int r = 0; r < 8; ++r) {
            if (rbase + r < nrows) op[(size_t)r * 128] = c[r];
        }
    }
}

// ---------- per-node attention scalars: ALS = XS.a_s, ALD = XD.a_d ----------
__global__ void __launch_bounds__(256)
gemv_al_k(const float* __restrict__ XS, const float* __restrict__ XD,
          const float* __restrict__ as_, const float* __restrict__ ad_,
          float* __restrict__ ALS, float* __restrict__ ALD, int n) {
    int gid  = blockIdx.x * blockDim.x + threadIdx.x;
    int node = gid >> 5, lane = gid & 31;
    if (node >= n) return;
    float4 av = *(const float4*)(as_ + lane * 4);
    float4 dv = *(const float4*)(ad_ + lane * 4);
    float4 xs = *(const float4*)(XS + (size_t)node * 128 + lane * 4);
    float4 xd = *(const float4*)(XD + (size_t)node * 128 + lane * 4);
    float s = xs.x * av.x + xs.y * av.y + xs.z * av.z + xs.w * av.w;
    float t = xd.x * dv.x + xd.y * dv.y + xd.z * dv.z + xd.w * dv.w;
#pragma unroll
    for (int off = 16; off; off >>= 1) {
        s += __shfl_xor(s, off, 32);
        t += __shfl_xor(t, off, 32);
    }
    if (lane == 0) { ALS[node] = s; ALD[node] = t; }
}

// ---------- clear accumulators ----------
__global__ void init_k(float* __restrict__ ACC, unsigned* __restrict__ NMAX,
                       float* __restrict__ NSUM, int nNodes) {
    int i = blockIdx.x * blockDim.x + threadIdx.x;
    if (i < nNodes * 128) ACC[i] = 0.f;
    if (i < nNodes) { NMAX[i] = 0u; NSUM[i] = 0.f; }
}

// ---------- edge logits + segment max ----------
__global__ void logits_k(const int* __restrict__ src, const int* __restrict__ dst,
                         const float* __restrict__ ALS, const float* __restrict__ ALD,
                         float* __restrict__ EW, unsigned* __restrict__ NMAX, int E) {
    int e = blockIdx.x * blockDim.x + threadIdx.x;
    if (e >= E) return;
    int d = dst[e];
    float l = ALS[src[e]] + ALD[d];
    l = (l > 0.f) ? l : 0.2f * l;                 // leaky_relu
    EW[e] = l;
    atomicMax(NMAX + d, encf(l));
}

// ---------- exp + segment sum ----------
__global__ void expsum_k(const int* __restrict__ dst, const unsigned* __restrict__ NMAX,
                         float* __restrict__ EW, float* __restrict__ NSUM, int E) {
    int e = blockIdx.x * blockDim.x + threadIdx.x;
    if (e >= E) return;
    int d = dst[e];
    float ex = __expf(EW[e] - decf(NMAX[d]));
    EW[e] = ex;
    atomicAdd(NSUM + d, ex);
}

// ---------- weighted scatter-add: one wave per edge ----------
__global__ void __launch_bounds__(256)
aggr_k(const int* __restrict__ src, const int* __restrict__ dst,
       const float* __restrict__ XS, const float* __restrict__ EW,
       const float* __restrict__ NSUM, float* __restrict__ ACC, int E) {
    int gid = blockIdx.x * blockDim.x + threadIdx.x;
    int e = gid >> 5, lane = gid & 31;
    if (e >= E) return;
    int s = src[e], d = dst[e];
    float alpha = EW[e] / (NSUM[d] + 1e-16f);
    float4 xv = *(const float4*)(XS + (size_t)s * 128 + lane * 4);
    float* o = ACC + (size_t)d * 128 + lane * 4;
    atomicAdd(o + 0, alpha * xv.x);
    atomicAdd(o + 1, alpha * xv.y);
    atomicAdd(o + 2, alpha * xv.z);
    atomicAdd(o + 3, alpha * xv.w);
}

// ---------- bias + relu fused into bf16 conversion (layer1 -> layer2 input) ----------
__global__ void bias_relu_bf16_k(const float* __restrict__ ACC, const float* __restrict__ b,
                                 unsigned short* __restrict__ XB, int n) {
    int i = blockIdx.x * blockDim.x + threadIdx.x;
    if (i >= n) return;
    float v = ACC[i] + b[i & 127];
    XB[i] = f2bf(v > 0.f ? v : 0.f);
}

// ---------- final bias add ----------
__global__ void final_k(const float* __restrict__ ACC, const float* __restrict__ b,
                        float* __restrict__ out, int n) {
    int i = blockIdx.x * blockDim.x + threadIdx.x;
    if (i < n) out[i] = ACC[i] + b[i & 127];
}

extern "C" void kernel_launch(void* const* d_in, const int* in_sizes, int n_in,
                              void* d_out, int out_size, void* d_ws, size_t ws_size,
                              hipStream_t stream) {
    const float* x   = (const float*)d_in[0];
    const int*   ei  = (const int*)d_in[1];
    const float* W1s = (const float*)d_in[2];
    const float* W1d = (const float*)d_in[3];
    const float* a1s = (const float*)d_in[4];
    const float* a1d = (const float*)d_in[5];
    const float* b1  = (const float*)d_in[6];
    const float* W2s = (const float*)d_in[7];
    const float* W2d = (const float*)d_in[8];
    const float* a2s = (const float*)d_in[9];
    const float* a2d = (const float*)d_in[10];
    const float* b2  = (const float*)d_in[11];

    const int N = in_sizes[0] / 128;
    const int E = in_sizes[1] / 2;
    const int* src = ei;
    const int* dst = ei + E;

    // workspace carve-out (256B aligned)
    char* p = (char*)d_ws;
    auto carve = [&](size_t bytes) -> char* {
        char* r = p;
        p += (bytes + 255) & ~(size_t)255;
        return r;
    };
    unsigned short* XB   = (unsigned short*)carve((size_t)N * 128 * 2);
    unsigned short* WP   = (unsigned short*)carve((size_t)4 * 128 * 128 * 2);
    float*          XS   = (float*)carve((size_t)N * 128 * 4);
    float*          XD   = (float*)carve((size_t)N * 128 * 4);
    float*          ACC  = (float*)carve((size_t)N * 128 * 4);
    float*          ALS  = (float*)carve((size_t)N * 4);
    float*          ALD  = (float*)carve((size_t)N * 4);
    unsigned*       NMAX = (unsigned*)carve((size_t)N * 4);
    float*          NSUM = (float*)carve((size_t)N * 4);
    float*          EW   = (float*)carve((size_t)E * 4);

    const int totalNF = N * 128;
    const int TPB = 256;

    pack_w_k<<<(4 * 128 * 128) / TPB, TPB, 0, stream>>>(W1s, W1d, W2s, W2d, WP);
    cvt_bf16_k<<<(totalNF + TPB - 1) / TPB, TPB, 0, stream>>>(x, XB, totalNF);

    for (int layer = 0; layer < 2; ++layer) {
        const float* as_ = layer ? a2s : a1s;
        const float* ad_ = layer ? a2d : a1d;
        const float* b_  = layer ? b2  : b1;

        dim3 gg((N + 15) / 16, 2);
        gemm_bf16_k<<<gg, TPB, 0, stream>>>(XB, WP, XS, XD, layer * 2, N);

        gemv_al_k<<<(N * 32 + TPB - 1) / TPB, TPB, 0, stream>>>(XS, XD, as_, ad_, ALS, ALD, N);

        init_k<<<(totalNF + TPB - 1) / TPB, TPB, 0, stream>>>(ACC, NMAX, NSUM, N);
        logits_k<<<(E + TPB - 1) / TPB, TPB, 0, stream>>>(src, dst, ALS, ALD, EW, NMAX, E);
        expsum_k<<<(E + TPB - 1) / TPB, TPB, 0, stream>>>(dst, NMAX, EW, NSUM, E);
        aggr_k<<<(int)(((long long)E * 32 + TPB - 1) / TPB), TPB, 0, stream>>>(
            src, dst, XS, EW, NSUM, ACC, E);

        if (layer == 0)
            bias_relu_bf16_k<<<(totalNF + TPB - 1) / TPB, TPB, 0, stream>>>(ACC, b_, XB, totalNF);
        else
            final_k<<<(totalNF + TPB - 1) / TPB, TPB, 0, stream>>>(ACC, b_, (float*)d_out, totalNF);
    }
}
